// SSMKernel_52733608460862
// MI455X (gfx1250) — matrix operand order repaired and checked by reading, not compiled, not run on hardware
//
#include <hip/hip_runtime.h>
#include <hip/hip_bf16.h>
#include <math.h>

typedef __attribute__((ext_vector_type(16))) _Float16 v16h;
typedef __attribute__((ext_vector_type(8)))  _Float16 h8;
typedef __attribute__((ext_vector_type(8)))  float    v8f;
typedef __attribute__((__vector_size__(16))) int      gv4i;   // builtin's pointee type

#define D_MODEL 768
#define D_STATE 16
#define SEQ_L   2048
#define BATCH   4
#define TOKENS  (BATCH * SEQ_L)   // 8192

// CDNA5 async global->LDS DMA path (gfx1250): guarded so the file compiles on
// toolchains that do not expose the builtin; fallback = sync staging.
#if defined(__has_builtin)
#  if __has_builtin(__builtin_amdgcn_global_load_async_to_lds_b128)
#    define HAVE_ASYNC_LDS 1
#  endif
#endif
#ifndef HAVE_ASYNC_LDS
#  define HAVE_ASYNC_LDS 0
#endif

template <int N>
__device__ __forceinline__ void wait_asynccnt_le() {
#if HAVE_ASYNC_LDS
#  if __has_builtin(__builtin_amdgcn_s_wait_asynccnt)
    __builtin_amdgcn_s_wait_asynccnt(N);
#  else
    asm volatile("s_wait_asynccnt %0" ::"i"(N) : "memory");
#  endif
#endif
}

__device__ __forceinline__ void stage_b128(const _Float16* src, _Float16* dst) {
#if HAVE_ASYNC_LDS
    __builtin_amdgcn_global_load_async_to_lds_b128((gv4i*)src, (gv4i*)dst, 0, 0);
#else
    *(h8*)dst = *(const h8*)src;
#endif
}

// ---------------------------------------------------------------------------
// Prep kernels: f32 -> f16 convert, and transposing convert for weights.
// ---------------------------------------------------------------------------
__global__ void cvt_f32_to_f16(const float* __restrict__ in,
                               _Float16* __restrict__ out, int n) {
    int i = blockIdx.x * blockDim.x + threadIdx.x;
    if (i < n) out[i] = (_Float16)in[i];
}

// in: [R, C] f32 row-major  ->  out: [C, R] f16 row-major
__global__ void transpose_cvt(const float* __restrict__ in,
                              _Float16* __restrict__ out, int R, int C) {
    int idx = blockIdx.x * blockDim.x + threadIdx.x;
    if (idx >= R * C) return;
    int i = idx / C;
    int j = idx - i * C;
    out[(size_t)j * R + i] = (_Float16)in[idx];
}

// ---------------------------------------------------------------------------
// Epilogue helper
// EPI: 0 = SiLU -> f16 store    1 = identity -> f32
//      2 = -|h| -> f32          3 = sigmoid -> f32
// ---------------------------------------------------------------------------
template <int EPI>
__device__ __forceinline__ void epi_store(void* out, size_t oidx, float h) {
    if (EPI == 0) {
        float s = h * (1.0f / (1.0f + __expf(-h)));
        ((_Float16*)out)[oidx] = (_Float16)s;
    } else if (EPI == 1) {
        ((float*)out)[oidx] = h;
    } else if (EPI == 2) {
        ((float*)out)[oidx] = -fabsf(h);
    } else {
        ((float*)out)[oidx] = 1.0f / (1.0f + __expf(-h));
    }
}

// ---------------------------------------------------------------------------
// Blocked WMMA GEMM: out[M,N] = epi(A[M,K] @ BT[N,K]^T + bias)
// Block = 256 threads = 8 waves. Block macro-tile: 256 rows x 64 cols.
// Wave tile: 32 rows x 64 cols = 2 m-tiles x 4 n-tiles, 8 accumulators.
// B tile (64 cols x 32 k, 4KB) double-buffered in LDS, filled by async
// global->LDS DMA (GLOBAL_LOAD_ASYNC_TO_LDS_B128) one K-step ahead.
// Requires M % 256 == 0, N % 64 == 0, K % 32 == 0.
// ---------------------------------------------------------------------------
template <int EPI>
__global__ __launch_bounds__(256)
void gemm_wmma_blocked(const _Float16* __restrict__ A,
                       const _Float16* __restrict__ BT,
                       const float* __restrict__ bias,
                       void* __restrict__ out,
                       int M, int N, int K) {
    __shared__ _Float16 sB[2][64 * 32];   // [buf][n_local][k_local]

    const int lane = threadIdx.x & 31;
    const int wave = threadIdx.x >> 5;
    const int sub  = lane & 15;        // row (A) / col (B,D) within 16-tile
    const int hi   = lane >> 4;        // K half-group carried by this lane

    const int nbase = blockIdx.x * 64;
    const int mbase = blockIdx.y * 256 + wave * 32;

    // A-fragment sources for the wave's two m-tiles
    const _Float16* pa0 = A + (size_t)(mbase + sub) * K + (hi << 4);
    const _Float16* pa1 = pa0 + (size_t)16 * K;

    // cooperative B-copy assignment: 64 rows x 4 x b128 chunks per K-step
    const int crow   = threadIdx.x >> 2;        // 0..63  (n_local)
    const int cchunk = (threadIdx.x & 3) << 3;  // 0,8,16,24 (halves)
    const _Float16* pbsrc = BT + (size_t)(nbase + crow) * K + cchunk;
    const int soff = crow * 32 + cchunk;

    const int nk = K >> 5;
    v8f acc[2][4] = {};

    // prologue: start filling buffer 0 with K-step 0
    stage_b128(pbsrc, &sB[0][soff]);

    for (int i = 0; i < nk; ++i) {
        const int cur = i & 1;
        if (i + 1 < nk) {
            stage_b128(pbsrc + ((i + 1) << 5), &sB[cur ^ 1][soff]);
            wait_asynccnt_le<1>();   // my K-step i copy has landed
        } else {
            wait_asynccnt_le<0>();
        }
        __syncthreads();             // everyone's K-step i copy has landed

        const int k0 = i << 5;
        h8 a0lo = *(const h8*)(pa0 + k0);
        h8 a0hi = *(const h8*)(pa0 + k0 + 8);
        h8 a1lo = *(const h8*)(pa1 + k0);
        h8 a1hi = *(const h8*)(pa1 + k0 + 8);
        v16h a0 = __builtin_shufflevector(a0lo, a0hi,
                    0,1,2,3,4,5,6,7,8,9,10,11,12,13,14,15);
        v16h a1 = __builtin_shufflevector(a1lo, a1hi,
                    0,1,2,3,4,5,6,7,8,9,10,11,12,13,14,15);

#pragma unroll
        for (int j = 0; j < 4; ++j) {
            v16h bf = *(const v16h*)(&sB[cur][(j * 16 + sub) * 32 + (hi << 4)]);
            acc[0][j] = __builtin_amdgcn_wmma_f32_16x16x32_f16(
                            false, a0, false, bf, (short)0, acc[0][j], false, false);
            acc[1][j] = __builtin_amdgcn_wmma_f32_16x16x32_f16(
                            false, a1, false, bf, (short)0, acc[1][j], false, false);
        }
        __syncthreads();             // reads of sB[cur] done -> free for i+2
    }

    // D layout: lane holds column (lane&15); VGPR v holds row v + 8*hi
#pragma unroll
    for (int mi = 0; mi < 2; ++mi) {
#pragma unroll
        for (int j = 0; j < 4; ++j) {
            const int col  = nbase + j * 16 + sub;
            const float bv = bias[col];
            const int rb   = mbase + mi * 16 + (hi << 3);
#pragma unroll
            for (int v = 0; v < 8; ++v) {
                epi_store<EPI>(out, (size_t)(rb + v) * N + col, acc[mi][j][v] + bv);
            }
        }
    }
}

// ---------------------------------------------------------------------------
// Simple WMMA GEMM (one 16x16 tile per wave) for the skinny N=16 layer-2s.
// ---------------------------------------------------------------------------
template <int EPI>
__global__ void gemm_wmma_f16(const _Float16* __restrict__ A,
                              const _Float16* __restrict__ BT,
                              const float* __restrict__ bias,
                              void* __restrict__ out,
                              int M, int N, int K) {
    const int lane = threadIdx.x & 31;
    const int wave = threadIdx.x >> 5;
    const int ntiles = N >> 4;
    const int total  = (M >> 4) * ntiles;
    int tile = blockIdx.x * (blockDim.x >> 5) + wave;
    if (tile >= total) return;
    const int mt = tile / ntiles;
    const int nt = tile - mt * ntiles;

    const int sub = lane & 15;
    const int hi  = lane >> 4;

    const _Float16* pa = A  + (size_t)((mt << 4) + sub) * K + (hi << 4);
    const _Float16* pb = BT + (size_t)((nt << 4) + sub) * K + (hi << 4);

    v8f acc = {};
    for (int k0 = 0; k0 < K; k0 += 32) {
        h8 a_lo = *(const h8*)(pa + k0);
        h8 a_hi = *(const h8*)(pa + k0 + 8);
        v16h af = __builtin_shufflevector(a_lo, a_hi,
                     0,1,2,3,4,5,6,7,8,9,10,11,12,13,14,15);
        v16h bf = *(const v16h*)(pb + k0);
        acc = __builtin_amdgcn_wmma_f32_16x16x32_f16(
                  false, af, false, bf, (short)0, acc, false, false);
    }

    const int col = (nt << 4) + sub;
    const float bv = bias[col];
    const int mrow = (mt << 4) + (hi << 3);
#pragma unroll
    for (int v = 0; v < 8; ++v) {
        epi_store<EPI>(out, (size_t)(mrow + v) * N + col, acc[v] + bv);
    }
}

// ---------------------------------------------------------------------------
// Sequential SSM scan. One thread per (b,d) channel, 16 states in registers.
// ---------------------------------------------------------------------------
__global__ void ssm_scan(const float* __restrict__ x,       // [B,L,D] f32
                         const float* __restrict__ Alog,    // [B,L,N]
                         const float* __restrict__ Bv,      // [B,L,N]
                         const float* __restrict__ Cv,      // [B,L,N]
                         const float* __restrict__ delta,   // [B,L,D]
                         const float* __restrict__ Dparam,  // [D]
                         const float* __restrict__ log_dt,  // [D]
                         float* __restrict__ out) {         // [B,L,D]
    __shared__ float sA[D_STATE], sB[D_STATE], sC[D_STATE];
    const int ch = blockIdx.x * blockDim.x + threadIdx.x;   // 0..B*D-1
    const int b  = ch / D_MODEL;
    const int d  = ch - b * D_MODEL;

    const float dt = __expf(log_dt[d]);
    const float Dp = Dparam[d];

    float st[D_STATE];
#pragma unroll
    for (int n = 0; n < D_STATE; ++n) st[n] = 0.0f;

    for (int t = 0; t < SEQ_L; ++t) {
        const size_t tok = (size_t)b * SEQ_L + t;
        __syncthreads();
        if (threadIdx.x < D_STATE) {
            sA[threadIdx.x] = Alog[tok * D_STATE + threadIdx.x];
            sB[threadIdx.x] = Bv  [tok * D_STATE + threadIdx.x];
            sC[threadIdx.x] = Cv  [tok * D_STATE + threadIdx.x];
        }
        __syncthreads();

        const float dlt = delta[tok * D_MODEL + d];
        const float u   = x    [tok * D_MODEL + d];
        const float om  = 1.0f - dlt;
        float y = 0.0f;
#pragma unroll
        for (int n = 0; n < D_STATE; ++n) {
            float dA  = __expf(sA[n] * dt);
            float cur = u * (sB[n] * dt);
            st[n] = (dlt * st[n] + om * cur) * dA;
            y += st[n] * sC[n];
        }
        out[tok * D_MODEL + d] = y + u * Dp;
    }
}

// ---------------------------------------------------------------------------
// Host launch
// ---------------------------------------------------------------------------
static inline size_t align_up(size_t v) { return (v + 255) & ~(size_t)255; }

extern "C" void kernel_launch(void* const* d_in, const int* in_sizes, int n_in,
                              void* d_out, int out_size, void* d_ws, size_t ws_size,
                              hipStream_t stream) {
    const float* x     = (const float*)d_in[0];
    const float* A_w1  = (const float*)d_in[1];
    const float* A_b1  = (const float*)d_in[2];
    const float* A_w2  = (const float*)d_in[3];
    const float* A_b2  = (const float*)d_in[4];
    const float* B_w1  = (const float*)d_in[5];
    const float* B_b1  = (const float*)d_in[6];
    const float* B_w2  = (const float*)d_in[7];
    const float* B_b2  = (const float*)d_in[8];
    const float* C_w1  = (const float*)d_in[9];
    const float* C_b1  = (const float*)d_in[10];
    const float* C_w2  = (const float*)d_in[11];
    const float* C_b2  = (const float*)d_in[12];
    const float* d_w1  = (const float*)d_in[13];
    const float* d_b1  = (const float*)d_in[14];
    const float* d_w2  = (const float*)d_in[15];
    const float* d_b2  = (const float*)d_in[16];
    const float* Dpar  = (const float*)d_in[17];
    const float* logdt = (const float*)d_in[18];
    float* out = (float*)d_out;

    char* ws = (char*)d_ws;
    size_t off = 0;
    auto take = [&](size_t bytes) { char* p = ws + off; off = align_up(off + bytes); return p; };

    _Float16* xf16  = (_Float16*)take((size_t)TOKENS * D_MODEL * 2);
    _Float16* w1T_A = (_Float16*)take((size_t)2 * D_MODEL * D_MODEL * 2); // [1536,768]
    _Float16* w1T_B = (_Float16*)take((size_t)2 * D_MODEL * D_MODEL * 2);
    _Float16* w1T_C = (_Float16*)take((size_t)2 * D_MODEL * D_MODEL * 2);
    _Float16* w1T_d = (_Float16*)take((size_t)D_MODEL * D_MODEL * 2);     // [768,768]
    _Float16* w2T_A = (_Float16*)take((size_t)D_STATE * 2 * D_MODEL * 2); // [16,1536]
    _Float16* w2T_B = (_Float16*)take((size_t)D_STATE * 2 * D_MODEL * 2);
    _Float16* w2T_C = (_Float16*)take((size_t)D_STATE * 2 * D_MODEL * 2);
    _Float16* w2T_d = (_Float16*)take((size_t)D_MODEL * D_MODEL * 2);
    _Float16* hbuf  = (_Float16*)take((size_t)TOKENS * 2 * D_MODEL * 2);  // [8192,1536]
    float*    outA  = (float*)take((size_t)TOKENS * D_STATE * 4);
    float*    outB  = (float*)take((size_t)TOKENS * D_STATE * 4);
    float*    outC  = (float*)take((size_t)TOKENS * D_STATE * 4);
    float*    deltaf= (float*)take((size_t)TOKENS * D_MODEL * 4);
    (void)ws_size; (void)in_sizes; (void)n_in; (void)out_size;

    const int TB = 256;
    // --- convert activations ---
    {
        int n = TOKENS * D_MODEL;
        cvt_f32_to_f16<<<(n + TB - 1) / TB, TB, 0, stream>>>(x, xf16, n);
    }
    // --- convert + transpose weights ---
    auto xpose = [&](const float* in, _Float16* o, int R, int C) {
        int n = R * C;
        transpose_cvt<<<(n + TB - 1) / TB, TB, 0, stream>>>(in, o, R, C);
    };
    xpose(A_w1, w1T_A, D_MODEL, 2 * D_MODEL);
    xpose(B_w1, w1T_B, D_MODEL, 2 * D_MODEL);
    xpose(C_w1, w1T_C, D_MODEL, 2 * D_MODEL);
    xpose(d_w1, w1T_d, D_MODEL, D_MODEL);
    xpose(A_w2, w2T_A, 2 * D_MODEL, D_STATE);
    xpose(B_w2, w2T_B, 2 * D_MODEL, D_STATE);
    xpose(C_w2, w2T_C, 2 * D_MODEL, D_STATE);
    xpose(d_w2, w2T_d, D_MODEL, D_MODEL);

    auto blk_grid = [&](int M, int N) { return dim3(N / 64, M / 256); };
    auto thin_grid = [&](int M, int N) {
        int tiles = (M >> 4) * (N >> 4);
        return (tiles + (TB >> 5) - 1) / (TB >> 5);
    };

    // --- A path: A_log = -|silu(x@w1+b1)@w2+b2| ---
    gemm_wmma_blocked<0><<<blk_grid(TOKENS, 2 * D_MODEL), TB, 0, stream>>>(
        xf16, w1T_A, A_b1, hbuf, TOKENS, 2 * D_MODEL, D_MODEL);
    gemm_wmma_f16<2><<<thin_grid(TOKENS, D_STATE), TB, 0, stream>>>(
        hbuf, w2T_A, A_b2, outA, TOKENS, D_STATE, 2 * D_MODEL);
    // --- B path ---
    gemm_wmma_blocked<0><<<blk_grid(TOKENS, 2 * D_MODEL), TB, 0, stream>>>(
        xf16, w1T_B, B_b1, hbuf, TOKENS, 2 * D_MODEL, D_MODEL);
    gemm_wmma_f16<1><<<thin_grid(TOKENS, D_STATE), TB, 0, stream>>>(
        hbuf, w2T_B, B_b2, outB, TOKENS, D_STATE, 2 * D_MODEL);
    // --- C path ---
    gemm_wmma_blocked<0><<<blk_grid(TOKENS, 2 * D_MODEL), TB, 0, stream>>>(
        xf16, w1T_C, C_b1, hbuf, TOKENS, 2 * D_MODEL, D_MODEL);
    gemm_wmma_f16<1><<<thin_grid(TOKENS, D_STATE), TB, 0, stream>>>(
        hbuf, w2T_C, C_b2, outC, TOKENS, D_STATE, 2 * D_MODEL);
    // --- delta path: sigmoid(silu(x@w1+b1)@w2+b2) ---
    gemm_wmma_blocked<0><<<blk_grid(TOKENS, D_MODEL), TB, 0, stream>>>(
        xf16, w1T_d, d_b1, hbuf, TOKENS, D_MODEL, D_MODEL);
    gemm_wmma_blocked<3><<<blk_grid(TOKENS, D_MODEL), TB, 0, stream>>>(
        hbuf, w2T_d, d_b2, deltaf, TOKENS, D_MODEL, D_MODEL);

    // --- sequential scan ---
    ssm_scan<<<(BATCH * D_MODEL) / TB, TB, 0, stream>>>(
        x, outA, outB, outC, deltaf, Dpar, logdt, out);
}
